// Multihead_mask_attention_77249281786479
// MI455X (gfx1250) — compile-verified, matrix operand order checked by reading, not run on hardware
//
#include <hip/hip_runtime.h>
#include <hip/hip_bf16.h>

// ---------------------------------------------------------------------------
// Multi-head causal attention for gfx1250 (MI455X), WMMA f16 path.
//   D_MODEL=1024, N_HEADS=16, D_K=64, B=4, S=2048  ->  M = B*S = 8192 tokens
// ---------------------------------------------------------------------------

typedef __attribute__((ext_vector_type(16))) _Float16 v16h;
typedef __attribute__((ext_vector_type(8)))  _Float16 v8h;
typedef __attribute__((ext_vector_type(8)))  float    v8f;

#define SEQ     2048
#define DM      1024
#define NH      16
#define DK      64
#define NBATCH  4
#define MTOK    (NBATCH * SEQ)   // 8192

__device__ __forceinline__ v16h make_a(v8h lo, v8h hi) {
  v16h a;
#pragma unroll
  for (int i = 0; i < 8; ++i) { a[i] = lo[i]; a[8 + i] = hi[i]; }
  return a;
}

__device__ __forceinline__ v8f wmma_f16(v16h a, v16h b, v8f c) {
  // D = A(16x32 f16) * B(32x16 f16) + C(16x16 f32)
  return __builtin_amdgcn_wmma_f32_16x16x32_f16(
      /*neg_a=*/false, a, /*neg_b=*/false, b,
      /*c_mod=*/(short)0, c, /*reuse_a=*/false, /*reuse_b=*/false);
}

__device__ __forceinline__ float rowmax16(float v) {
#pragma unroll
  for (int m = 1; m < 16; m <<= 1) v = fmaxf(v, __shfl_xor(v, m, 32));
  return v;
}
__device__ __forceinline__ float rowsum16(float v) {
#pragma unroll
  for (int m = 1; m < 16; m <<= 1) v += __shfl_xor(v, m, 32);
  return v;
}

// ---------------------------------------------------------------------------
// f32 -> f16 conversion (grid-stride)
// ---------------------------------------------------------------------------
__global__ void cvt_f32_f16(const float* __restrict__ s, _Float16* __restrict__ d, int n) {
  int i = blockIdx.x * blockDim.x + threadIdx.x;
  int stride = gridDim.x * blockDim.x;
  for (; i < n; i += stride) d[i] = (_Float16)s[i];
}

// ---------------------------------------------------------------------------
// C[M,N] = A[M,K](f16) @ W[N,K]^T (f16) + bias.
// Register-blocked: one wave computes a 32x64 C strip (2 A-frags x 4 B-frags
// -> 8 WMMAs per 32-wide k-step; 768 B of fragment traffic per WMMA).
// A-fragment: lane holds row (lane&15), k-chunks at kbase=(lane>>4)*8, +16.
// B-fragment: lane holds col (lane&15); B[k][n]=W[n][k] -> contiguous W row.
// ---------------------------------------------------------------------------
template <bool F16OUT>
__global__ __launch_bounds__(256) void gemm_xwT(
    const _Float16* __restrict__ A, const _Float16* __restrict__ W,
    const float* __restrict__ bias, _Float16* __restrict__ outH,
    float* __restrict__ outF, int M, int N, int K) {
  const int wid   = threadIdx.x >> 5;
  const int lane  = threadIdx.x & 31;
  const int lrow  = lane & 15;
  const int lh8   = (lane >> 4) * 8;

  const long tile = (long)blockIdx.x * 8 + wid;   // 32x64 strips
  const int ntn = N >> 6;                         // strips along N
  const int mt = (int)(tile / ntn), nt = (int)(tile % ntn);
  const int m0 = mt << 5, n0 = nt << 6;

  const _Float16* ar0 = A + (size_t)(m0 + lrow) * K + lh8;
  const _Float16* ar1 = A + (size_t)(m0 + 16 + lrow) * K + lh8;
  const _Float16* br0 = W + (size_t)(n0 + lrow) * K + lh8;
  const _Float16* br1 = W + (size_t)(n0 + 16 + lrow) * K + lh8;
  const _Float16* br2 = W + (size_t)(n0 + 32 + lrow) * K + lh8;
  const _Float16* br3 = W + (size_t)(n0 + 48 + lrow) * K + lh8;

  v8f acc[2][4];
#pragma unroll
  for (int i = 0; i < 2; ++i)
#pragma unroll
    for (int j = 0; j < 4; ++j) acc[i][j] = (v8f){};

  for (int k0 = 0; k0 < K; k0 += 32) {
    const v16h a0 = make_a(*(const v8h*)(ar0 + k0), *(const v8h*)(ar0 + k0 + 16));
    const v16h a1 = make_a(*(const v8h*)(ar1 + k0), *(const v8h*)(ar1 + k0 + 16));
    const v16h b0 = make_a(*(const v8h*)(br0 + k0), *(const v8h*)(br0 + k0 + 16));
    const v16h b1 = make_a(*(const v8h*)(br1 + k0), *(const v8h*)(br1 + k0 + 16));
    const v16h b2 = make_a(*(const v8h*)(br2 + k0), *(const v8h*)(br2 + k0 + 16));
    const v16h b3 = make_a(*(const v8h*)(br3 + k0), *(const v8h*)(br3 + k0 + 16));
    acc[0][0] = wmma_f16(a0, b0, acc[0][0]);
    acc[0][1] = wmma_f16(a0, b1, acc[0][1]);
    acc[0][2] = wmma_f16(a0, b2, acc[0][2]);
    acc[0][3] = wmma_f16(a0, b3, acc[0][3]);
    acc[1][0] = wmma_f16(a1, b0, acc[1][0]);
    acc[1][1] = wmma_f16(a1, b1, acc[1][1]);
    acc[1][2] = wmma_f16(a1, b2, acc[1][2]);
    acc[1][3] = wmma_f16(a1, b3, acc[1][3]);
  }

#pragma unroll
  for (int j = 0; j < 4; ++j) {
    const int col = n0 + j * 16 + lrow;
    const float bv = bias[col];
#pragma unroll
    for (int i = 0; i < 2; ++i) {
#pragma unroll
      for (int r = 0; r < 8; ++r) {
        const int row = m0 + i * 16 + r + lh8;
        const float v = acc[i][j][r] + bv;
        if (F16OUT) outH[(size_t)row * N + col] = (_Float16)v;
        else        outF[(size_t)row * N + col] = v;
      }
    }
  }
}

// ---------------------------------------------------------------------------
// Flash attention: grid = (B*NH, SEQ/64); block = 128 (4 waves).
// Wave w owns q-rows [qbase + 16w, +16). kv tiles of 32 staged through LDS.
// ---------------------------------------------------------------------------
__global__ __launch_bounds__(128) void attn_kernel(
    const _Float16* __restrict__ Q, const _Float16* __restrict__ Kx,
    const _Float16* __restrict__ V, _Float16* __restrict__ O) {
  __shared__ __align__(16) _Float16 ksh[32 * 64];      // [kvrow][dk]
  __shared__ __align__(16) _Float16 vsh[64 * 32];      // transposed: [dk][kvrow]
  __shared__ __align__(16) _Float16 psh[4][16 * 32];   // per-wave P staging

  const int bh = blockIdx.x;
  const int b = bh >> 4, h = bh & 15;
  const int qbase = blockIdx.y * 64;
  const int wid  = threadIdx.x >> 5;
  const int lane = threadIdx.x & 31;
  const int lrow = lane & 15;
  const int lh8  = (lane >> 4) * 8;
  const int qstart = qbase + wid * 16;

  const size_t headBase = (size_t)b * SEQ * DM + (size_t)h * DK;

  // Q fragments for this wave's 16 rows: dk 0..31 and dk 32..63
  const _Float16* qrow = Q + headBase + (size_t)(qstart + lrow) * DM + lh8;
  const v16h qa0 = make_a(*(const v8h*)(qrow),      *(const v8h*)(qrow + 16));
  const v16h qa1 = make_a(*(const v8h*)(qrow + 32), *(const v8h*)(qrow + 48));

  float m[8], l[8];
  v8f o0 = {}, o1 = {}, o2 = {}, o3 = {};
#pragma unroll
  for (int r = 0; r < 8; ++r) { m[r] = -3.0e38f; l[r] = 0.0f; }

  const int ntiles = (qbase + 64) >> 5;   // causal: kv <= qbase+63
  for (int t = 0; t < ntiles; ++t) {
    const int kvb = t * 32;

    // ---- cooperative K/V tile load (128 threads: 32 rows x 4 chunks) ----
    {
      const int r = threadIdx.x >> 2;
      const int c = (threadIdx.x & 3) * 16;
      const _Float16* kg = Kx + headBase + (size_t)(kvb + r) * DM + c;
      *(v8h*)&ksh[r * 64 + c]     = *(const v8h*)kg;
      *(v8h*)&ksh[r * 64 + c + 8] = *(const v8h*)(kg + 8);
      const _Float16* vg = V + headBase + (size_t)(kvb + r) * DM + c;
      v8h va = *(const v8h*)vg;
      v8h vb = *(const v8h*)(vg + 8);
#pragma unroll
      for (int i = 0; i < 8; ++i) {
        vsh[(c + i) * 32 + r]     = va[i];
        vsh[(c + 8 + i) * 32 + r] = vb[i];
      }
    }
    __syncthreads();

    // ---- S = Q @ K^T : two 16x16 tiles (kv cols 0..15 and 16..31) ----
    const _Float16* kr0 = &ksh[lrow * 64 + lh8];
    const _Float16* kr1 = &ksh[(lrow + 16) * 64 + lh8];
    v8f s0 = {}, s1 = {};
    s0 = wmma_f16(qa0, make_a(*(const v8h*)kr0,        *(const v8h*)(kr0 + 16)), s0);
    s0 = wmma_f16(qa1, make_a(*(const v8h*)(kr0 + 32), *(const v8h*)(kr0 + 48)), s0);
    s1 = wmma_f16(qa0, make_a(*(const v8h*)kr1,        *(const v8h*)(kr1 + 16)), s1);
    s1 = wmma_f16(qa1, make_a(*(const v8h*)(kr1 + 32), *(const v8h*)(kr1 + 48)), s1);

    // ---- causal mask, scale, online softmax update ----
    float p0f[8], p1f[8];
#pragma unroll
    for (int r = 0; r < 8; ++r) {
      const int q  = qstart + r + lh8;
      const int c0 = kvb + lrow, c1 = c0 + 16;
      const float x0 = (c0 <= q) ? s0[r] * 0.125f : -3.0e38f;  // 1/sqrt(64)
      const float x1 = (c1 <= q) ? s1[r] * 0.125f : -3.0e38f;
      const float mt = rowmax16(fmaxf(x0, x1));
      const float mn = fmaxf(m[r], mt);
      const float al = __expf(m[r] - mn);
      const float p0 = __expf(x0 - mn);
      const float p1 = __expf(x1 - mn);
      l[r] = l[r] * al + rowsum16(p0 + p1);
      m[r] = mn;
      o0[r] *= al; o1[r] *= al; o2[r] *= al; o3[r] *= al;
      p0f[r] = p0; p1f[r] = p1;
    }

    // ---- re-layout P (C-layout -> A-fragment) via per-wave LDS region ----
    _Float16* pw = &psh[wid][0];
#pragma unroll
    for (int r = 0; r < 8; ++r) {
      const int row = r + lh8;
      pw[row * 32 + lrow]      = (_Float16)p0f[r];
      pw[row * 32 + 16 + lrow] = (_Float16)p1f[r];
    }
    // DS ops are in-order within a wave: no barrier needed for this readback.
    const _Float16* pr = &pw[lrow * 32 + lh8];
    const v16h pa = make_a(*(const v8h*)pr, *(const v8h*)(pr + 16));

    // ---- O += P @ V  (V transposed in LDS -> contiguous B fragments) ----
    const _Float16* vr = &vsh[lrow * 32 + lh8];
    o0 = wmma_f16(pa, make_a(*(const v8h*)(vr +  0 * 512), *(const v8h*)(vr +  0 * 512 + 16)), o0);
    o1 = wmma_f16(pa, make_a(*(const v8h*)(vr +  1 * 512), *(const v8h*)(vr +  1 * 512 + 16)), o1);
    o2 = wmma_f16(pa, make_a(*(const v8h*)(vr +  2 * 512), *(const v8h*)(vr +  2 * 512 + 16)), o2);
    o3 = wmma_f16(pa, make_a(*(const v8h*)(vr +  3 * 512), *(const v8h*)(vr +  3 * 512 + 16)), o3);

    __syncthreads();   // before next tile overwrites ksh/vsh
  }

  // ---- normalize and store to attn-out [b, s, h*64 + dk] (f16) ----
#pragma unroll
  for (int r = 0; r < 8; ++r) {
    const int q = qstart + r + lh8;
    const float inv = (l[r] > 0.0f) ? 1.0f / l[r] : 0.0f;
    _Float16* ob = O + headBase + (size_t)q * DM;
    ob[ 0 + lrow] = (_Float16)(o0[r] * inv);
    ob[16 + lrow] = (_Float16)(o1[r] * inv);
    ob[32 + lrow] = (_Float16)(o2[r] * inv);
    ob[48 + lrow] = (_Float16)(o3[r] * inv);
  }
}

// ---------------------------------------------------------------------------
// Host-side launch
// ---------------------------------------------------------------------------
extern "C" void kernel_launch(void* const* d_in, const int* in_sizes, int n_in,
                              void* d_out, int out_size, void* d_ws, size_t ws_size,
                              hipStream_t stream) {
  (void)in_sizes; (void)n_in; (void)out_size; (void)ws_size;
  const float* x  = (const float*)d_in[0];
  // d_in[1] = mask (causal, computed analytically -> unused)
  const float* Wq = (const float*)d_in[2];
  const float* bq = (const float*)d_in[3];
  const float* Wk = (const float*)d_in[4];
  const float* bk = (const float*)d_in[5];
  const float* Wv = (const float*)d_in[6];
  const float* bv = (const float*)d_in[7];
  const float* Wo = (const float*)d_in[8];
  const float* bo = (const float*)d_in[9];
  float* out = (float*)d_out;

  // Workspace carve-up (halves)
  _Float16* ws  = (_Float16*)d_ws;
  _Float16* xh  = ws;                                   // 8192*1024
  _Float16* Wqh = xh  + (size_t)MTOK * DM;              // 1024*1024
  _Float16* Wkh = Wqh + (size_t)DM * DM;
  _Float16* Wvh = Wkh + (size_t)DM * DM;
  _Float16* Woh = Wvh + (size_t)DM * DM;
  _Float16* Qh  = Woh + (size_t)DM * DM;                // 8192*1024
  _Float16* Kh  = Qh  + (size_t)MTOK * DM;
  _Float16* Vh  = Kh  + (size_t)MTOK * DM;
  _Float16* Ah  = Vh  + (size_t)MTOK * DM;

  // 1) f32 -> f16 conversions
  cvt_f32_f16<<<1024, 256, 0, stream>>>(x,  xh,  MTOK * DM);
  cvt_f32_f16<<<512,  256, 0, stream>>>(Wq, Wqh, DM * DM);
  cvt_f32_f16<<<512,  256, 0, stream>>>(Wk, Wkh, DM * DM);
  cvt_f32_f16<<<512,  256, 0, stream>>>(Wv, Wvh, DM * DM);
  cvt_f32_f16<<<512,  256, 0, stream>>>(Wo, Woh, DM * DM);

  // 2) QKV projections: (M/32)*(N/64)=4096 wave-strips / 8 waves per block
  const int gemmBlocks = (MTOK / 32) * (DM / 64) / 8;   // 512
  gemm_xwT<true><<<gemmBlocks, 256, 0, stream>>>(xh, Wqh, bq, Qh, nullptr, MTOK, DM, DM);
  gemm_xwT<true><<<gemmBlocks, 256, 0, stream>>>(xh, Wkh, bk, Kh, nullptr, MTOK, DM, DM);
  gemm_xwT<true><<<gemmBlocks, 256, 0, stream>>>(xh, Wvh, bv, Vh, nullptr, MTOK, DM, DM);

  // 3) Causal flash attention: (B*NH) x (SEQ/64) blocks of 4 waves
  attn_kernel<<<dim3(NBATCH * NH, SEQ / 64), 128, 0, stream>>>(Qh, Kh, Vh, Ah);

  // 4) Output projection (f32 + bias -> d_out)
  gemm_xwT<false><<<gemmBlocks, 256, 0, stream>>>(Ah, Woh, bo, nullptr, out, MTOK, DM, DM);
}
